// filter_ConvAttention_23218593202898
// MI455X (gfx1250) — matrix operand order repaired and checked by reading, not compiled, not run on hardware
//
#include <hip/hip_runtime.h>

typedef __attribute__((ext_vector_type(16))) __bf16          v16bf;
typedef __attribute__((ext_vector_type(16))) unsigned short  v16us;
typedef __attribute__((ext_vector_type(8)))  float           v8f;
typedef __attribute__((ext_vector_type(4)))  unsigned        v4u;
typedef __attribute__((ext_vector_type(8)))  unsigned        v8u;

#define B_   8
#define CIN_ 256
#define H_   64
#define W_   64
#define S_   4096      // H*W
#define PW_  66        // padded H/W
#define KC_  2304      // CIN*9
#define NH_  8
#define DH_  32

__device__ __forceinline__ unsigned short f2bf(float f) {
  union { float f; unsigned int u; } x; x.f = f;
  unsigned int u = x.u;
  u += 0x7FFFu + ((u >> 16) & 1u);          // round-to-nearest-even
  return (unsigned short)(u >> 16);
}

// Raw LDS byte offset of a __shared__ object (generic -> addrspace(3) cast).
__device__ __forceinline__ unsigned lds_off(const void* p) {
  return (unsigned)(unsigned long long)(const __attribute__((address_space(3))) void*)p;
}

// Async DMA: 16 bytes per lane, global -> LDS, tracked by ASYNCcnt.
__device__ __forceinline__ void async_cp16(unsigned lds_byte_off, const void* gsrc) {
  asm volatile("global_load_async_to_lds_b128 %0, %1, off"
               :: "v"(lds_byte_off), "v"(gsrc) : "memory");
}
__device__ __forceinline__ void wait_async0() {
  asm volatile("s_wait_asynccnt 0" ::: "memory");
}

// Tensor Data Mover: 2D tile load, global -> LDS, tracked by TENSORcnt.
// D# per CDNA5 ISA ch.8: group0 = {flags, lds_addr, global_addr, type=2},
// group1 = {mask/data_size, dims, tile dims, strides}. VADDR2/3 omitted (2D).
// Element size fixed to 2 bytes (bf16). Strides in elements.
__device__ __forceinline__ void tdm_load_2d(unsigned lds_addr, const void* gaddr,
                                            unsigned tile_d0, unsigned tile_d1,
                                            unsigned d0_stride) {
  unsigned long long ga = (unsigned long long)gaddr;
  v4u g0;
  g0[0] = 1u;                                   // count=1, user descriptor
  g0[1] = lds_addr;                             // LDS byte address
  g0[2] = (unsigned)ga;                         // global_addr[31:0]
  g0[3] = (unsigned)(ga >> 32) | 0x80000000u;   // global_addr[56:32] | type=2
  v8u g1;
  g1[0] = 0x00010000u;                          // wg_mask=0, data_size=1 (2B)
  g1[1] = (tile_d0 & 0xFFFFu) << 16;            // tensor_dim0[15:0] (= tile)
  g1[2] = ((tile_d0 >> 16) & 0xFFFFu) | ((tile_d1 & 0xFFFFu) << 16); // td0 hi | td1 lo
  g1[3] = ((tile_d1 >> 16) & 0xFFFFu) | (tile_d0 << 16);             // td1 hi | tile_dim0
  g1[4] = tile_d1 & 0xFFFFu;                    // tile_dim1 (tile_dim2 = 0)
  g1[5] = d0_stride;                            // tensor_dim0_stride[31:0]
  g1[6] = 0u;                                   // d0_stride[47:32]=0, d1_stride[15:0]=0
  g1[7] = 0u;                                   // d1_stride[47:16]=0 (2D: unused)
  asm volatile("tensor_load_to_lds %0, %1" :: "s"(g0), "s"(g1) : "memory");
}

// Swizzled fused-weight index: Wfs[c0t(8)][mtile(16)][tap(9)][lane(32)][elem(16)]
// such that a lane's 16 A-fragment elements are 32 contiguous bytes, matching
// the B operand's contiguous K order (lanes 0-15: K=0-15; lanes 16-31: K=16-31).
__device__ __forceinline__ int swz_widx(int m, int j) {
  int cin = j / 9, tap = j - cin * 9;
  int c0t = cin >> 5, kl = cin & 31;
  int jj    = kl & 15;                        // element slot within fragment
  int lhalf = (kl >> 4) & 1;
  int lane  = (lhalf << 4) | (m & 15);
  int mtg   = m >> 4;
  return ((((c0t * 16 + mtg) * 9 + tap) * 32 + lane) << 4) + jj;
}

// ---------------------------------------------------------------------------
// scale[which*256+o] = g[o] / ||v[o,:]||  for q(512) / k(256) / v(256)
// ---------------------------------------------------------------------------
__global__ void scale_kernel(const float* __restrict__ qv, const float* __restrict__ qg,
                             const float* __restrict__ kv, const float* __restrict__ kg,
                             const float* __restrict__ vv, const float* __restrict__ vg,
                             float* __restrict__ scale) {
  int i = blockIdx.x * blockDim.x + threadIdx.x;
  if (i >= 768) return;
  int which = i >> 8, o = i & 255;
  const float* v; const float* g; int len;
  if (which == 0)      { v = qv; g = qg; len = 512; }
  else if (which == 1) { v = kv; g = kg; len = 256; }
  else                 { v = vv; g = vg; len = 256; }
  float s = 0.f;
  for (int c = 0; c < len; ++c) { float t = v[o * len + c]; s += t * t; }
  scale[i] = g[o] * rsqrtf(s);
}

// ---------------------------------------------------------------------------
// Fused conv weights, written in WMMA-fragment-swizzled order (bf16).
// ---------------------------------------------------------------------------
__global__ void fuse_qk_kernel(const float* __restrict__ linv, const float* __restrict__ scale,
                               const float* __restrict__ wconv, int clen,
                               unsigned short* __restrict__ Wf) {
  int idx = blockIdx.x * blockDim.x + threadIdx.x;          // 256*2304
  if (idx >= 256 * KC_) return;
  int o = idx / KC_, j = idx - o * KC_;
  float acc = 0.f;
  for (int c = 0; c < clen; ++c)
    acc += linv[o * clen + c] * wconv[c * KC_ + j];
  Wf[swz_widx(o, j)] = f2bf(acc * scale[o]);
}

// Grouped (groups=2) value conv fused with linear (block-diagonal conv weight).
__global__ void fuse_v_kernel(const float* __restrict__ linv, const float* __restrict__ scale,
                              const float* __restrict__ wconv,   // [256][128*9]
                              unsigned short* __restrict__ Wf) {
  int idx = blockIdx.x * blockDim.x + threadIdx.x;          // 256*2304
  if (idx >= 256 * KC_) return;
  int o = idx / KC_, j = idx - o * KC_;
  int cin = j / 9, tap = j - cin * 9;
  int grp = cin >> 7, lo = cin & 127;
  float acc = 0.f;
  for (int u = 0; u < 128; ++u)
    acc += linv[o * 256 + grp * 128 + u] * wconv[(grp * 128 + u) * 1152 + lo * 9 + tap];
  Wf[swz_widx(o, j)] = f2bf(acc * scale[o]);
}

// ---------------------------------------------------------------------------
// Concatenate left/right, convert to bf16, bake 1-pixel halo, channel-fastest:
// xpadT[b][r(66)][cc(66)][c(256)]. Coalesced writes; strided reads (one-shot).
// ---------------------------------------------------------------------------
__global__ void pad_kernel(const float* __restrict__ left, const float* __restrict__ right,
                           unsigned short* __restrict__ xpadT) {
  int idx = blockIdx.x * blockDim.x + threadIdx.x;
  const int total = B_ * PW_ * PW_ * CIN_;
  if (idx >= total) return;
  int c  = idx & 255;
  int t  = idx >> 8;
  int cc = t % PW_;
  int t2 = t / PW_;
  int r  = t2 % PW_;
  int b  = t2 / PW_;
  int h = r - 1, w = cc - 1;
  float val = 0.f;
  if ((unsigned)h < 64u && (unsigned)w < 64u) {
    if (c < 128) val = left [((b * 128 + c)         * 64 + h) * 64 + w];
    else         val = right[((b * 128 + (c - 128)) * 64 + h) * 64 + w];
  }
  xpadT[idx] = f2bf(val);
}

// ---------------------------------------------------------------------------
// Implicit-GEMM 3x3 conv, bf16 WMMA, f32 accumulate, double-buffered LDS
// (121.5 KB/workgroup; CDNA5 WGP has 320 KB). Software-pipelined staging:
// A slab via GLOBAL_LOAD_ASYNC_TO_LDS_B128 (ASYNCcnt), X slab via the
// Tensor Data Mover (TENSOR_LOAD_TO_LDS, TENSORcnt), overlapped under WMMA.
// O[256][B*S] = Wf[256][2304] x im2col[2304][B*S]
// Block: M=64 x N=256 (one batch, 4 image rows). 8 waves = 2(M) x 4(N);
// each wave: 32x64 = 2x4 WMMA tiles. K loop: 8 cin-tiles x 9 taps.
// ---------------------------------------------------------------------------
__global__ __launch_bounds__(256) void conv_wmma_kernel(
    const unsigned short* __restrict__ Wfs,    // swizzled [8][16][9][32][16] bf16
    const unsigned short* __restrict__ xpadT,  // [8][66][66][256] bf16
    const float* __restrict__ bias,            // [256]
    float* __restrict__ outbuf) {              // [B*S][256] f32
  __shared__ alignas(64) unsigned short lAb[2][4 * 9 * 32 * 16];  // 2 x 36 KB
  __shared__ alignas(64) unsigned short lXb[2][6 * 66 * 32];      // 2 x 25 KB

  const int tid   = threadIdx.x;
  const int lane  = tid & 31;
  const int wave  = tid >> 5;
  const int wm    = wave >> 2;                 // 0..1
  const int wn    = wave & 3;                  // 0..3
  const int lhalf = lane >> 4;                 // 0/1
  const int l16   = lane & 15;

  const int b    = blockIdx.x >> 4;            // batch
  const int h0   = (blockIdx.x & 15) << 2;     // first of 4 image rows
  const int m0   = blockIdx.y << 6;            // M tile origin
  const int mtg0 = blockIdx.y << 2;            // first global 16-row m tile

  // Issue all staging DMAs for cin-tile c0t into LDS buffer bufi.
  auto stage_slab = [&](int c0t, int bufi) {
    // A slab (async engine): 4 mtiles x 9 taps x 32 lanes x 16B = 4608 chunks
    unsigned laoff = lds_off(lAb[bufi]);
    const uint4* srcA = (const uint4*)(Wfs + (size_t)(c0t * 16 + mtg0) * 4608);
    #pragma unroll
    for (int i = 0; i < 18; ++i)
      async_cp16(laoff + ((unsigned)(tid + 256 * i) << 4), srcA + tid + 256 * i);
    // X slab (TDM): 2D tile = 396 rows (6 padded rows x 66 cols) of 32 bf16,
    // row pitch 256 elements. One descriptor, issued by wave 0 only.
    if (wave == 0) {
      const unsigned short* srcX = xpadT +
          (((size_t)(b * PW_ + h0) * PW_) << 8) + (c0t << 5);
      tdm_load_2d(lds_off(lXb[bufi]), srcX, /*tile_d0=*/32, /*tile_d1=*/396,
                  /*d0_stride=*/256);
    }
  };

  v8f acc[2][4];
  #pragma unroll
  for (int i = 0; i < 2; ++i)
    #pragma unroll
    for (int j = 0; j < 4; ++j)
      acc[i][j] = (v8f){0.f,0.f,0.f,0.f,0.f,0.f,0.f,0.f};

  stage_slab(0, 0);        // prologue: fill buffer 0
  wait_async0();
  __builtin_amdgcn_s_wait_tensorcnt(0);
  __syncthreads();

  for (int c0t = 0; c0t < 8; ++c0t) {
    const int cur = c0t & 1;
    if (c0t < 7) stage_slab(c0t + 1, cur ^ 1);   // DMAs for next slab in flight
    if (c0t < 6)   // prefetch the slab after that (global_prefetch_b8)
      __builtin_prefetch(Wfs + (size_t)((c0t + 2) * 16 + mtg0) * 4608, 0, 1);

    const unsigned short* lA = lAb[cur];
    const unsigned short* lX = lXb[cur];

    #pragma unroll
    for (int tap = 0; tap < 9; ++tap) {
      const int kh = tap / 3, kw = tap % 3;
      v16bf afrag[2];
      #pragma unroll
      for (int mt = 0; mt < 2; ++mt) {
        int mtl = (wm << 1) + mt;
        const v16us* p = (const v16us*)(lA + (((mtl * 9 + tap) * 32 + lane) << 4));
        afrag[mt] = __builtin_bit_cast(v16bf, *p);
      }
      v16bf bfrag[4];
      #pragma unroll
      for (int nt = 0; nt < 4; ++nt) {
        int nloc = (wn << 6) + (nt << 4) + l16;   // 0..255 within N tile
        int hl = nloc >> 6, w = nloc & 63;
        const v16us* p = (const v16us*)(lX +
            ((((hl + kh) * 66 + (w + kw)) << 5) + (lhalf << 4)));
        bfrag[nt] = __builtin_bit_cast(v16bf, *p);
      }
      #pragma unroll
      for (int mt = 0; mt < 2; ++mt)
        #pragma unroll
        for (int nt = 0; nt < 4; ++nt)
          acc[mt][nt] = __builtin_amdgcn_wmma_f32_16x16x32_bf16(
              false, afrag[mt], false, bfrag[nt], (short)0, acc[mt][nt], false, false);
    }

    wait_async0();                         // next A slab landed
    __builtin_amdgcn_s_wait_tensorcnt(0);  // next X slab landed (wave 0; free elsewhere)
    __syncthreads();                       // visible to all; fences buffer reuse
  }

  // Epilogue: D 16x16 f32 layout: elem r -> M = r + 8*(lane/16), N = lane%16.
  // Each 128B output line is fully written by one block -> L2 write-combines.
  const int nbase = b * S_ + h0 * 64;
  #pragma unroll
  for (int mt = 0; mt < 2; ++mt) {
    #pragma unroll
    for (int nt = 0; nt < 4; ++nt) {
      int nloc = (wn << 6) + (nt << 4) + l16;
      #pragma unroll
      for (int r = 0; r < 8; ++r) {
        int m = m0 + (wm << 5) + (mt << 4) + (lhalf << 3) + r;
        outbuf[(size_t)(nbase + nloc) * 256 + m] = acc[mt][nt][r] + bias[m];
      }
    }
  }
}

// ---------------------------------------------------------------------------
// Cross-head attention per position + layout scramble to NCHW.
// One thread per (b, s, head): attn[8] = softmax(q.k/sqrt(32)); out = attn @ v.
// ---------------------------------------------------------------------------
__global__ __launch_bounds__(256) void attn_kernel(
    const float* __restrict__ qbuf, const float* __restrict__ kbuf,
    const float* __restrict__ vbuf, float* __restrict__ out) {
  int idx = blockIdx.x * blockDim.x + threadIdx.x;  // B*S*NH = 262144
  int nh = idx & 7;
  int s  = (idx >> 3) & (S_ - 1);
  int b  = idx >> 15;
  const float* qp = qbuf + (size_t)(b * S_ + s) * 256 + nh * DH_;
  const float* kp = kbuf + (size_t)(b * S_ + s) * 256;
  const float* vp = vbuf + (size_t)(b * S_ + s) * 256;

  float q[DH_];
  #pragma unroll
  for (int d = 0; d < DH_; ++d) q[d] = qp[d];

  float sc[NH_];
  float mx = -1e30f;
  #pragma unroll
  for (int j = 0; j < NH_; ++j) {
    float a = 0.f;
    #pragma unroll
    for (int d = 0; d < DH_; ++d) a += q[d] * kp[j * DH_ + d];
    a *= 0.17677669529663687f;               // 1/sqrt(32)
    sc[j] = a;
    mx = fmaxf(mx, a);
  }
  float sum = 0.f;
  #pragma unroll
  for (int j = 0; j < NH_; ++j) { sc[j] = __expf(sc[j] - mx); sum += sc[j]; }
  float inv = 1.f / sum;

  #pragma unroll
  for (int d = 0; d < DH_; ++d) {
    float a = 0.f;
    #pragma unroll
    for (int j = 0; j < NH_; ++j) a += sc[j] * vp[j * DH_ + d];
    a *= inv;
    int L  = nh * (S_ * DH_) + s * DH_ + d;   // scramble: transpose(1,2)+reshape
    int c  = L & 255;
    int hw = L >> 8;                          // h*64 + w
    out[((size_t)(b * 256 + c) << 12) + hw] = a;
  }
}

// ---------------------------------------------------------------------------
extern "C" void kernel_launch(void* const* d_in, const int* in_sizes, int n_in,
                              void* d_out, int out_size, void* d_ws, size_t ws_size,
                              hipStream_t stream) {
  (void)in_sizes; (void)n_in; (void)out_size; (void)ws_size;
  const float* left  = (const float*)d_in[0];
  const float* right = (const float*)d_in[1];
  const float* wq    = (const float*)d_in[2];
  const float* wk    = (const float*)d_in[3];
  const float* wv    = (const float*)d_in[4];
  const float* q_v   = (const float*)d_in[5];
  const float* q_g   = (const float*)d_in[6];
  const float* q_b   = (const float*)d_in[7];
  const float* k_v   = (const float*)d_in[8];
  const float* k_g   = (const float*)d_in[9];
  const float* k_b   = (const float*)d_in[10];
  const float* v_v   = (const float*)d_in[11];
  const float* v_g   = (const float*)d_in[12];
  const float* v_b   = (const float*)d_in[13];
  float* out = (float*)d_out;

  char* ws = (char*)d_ws;
  size_t off = 0;
  auto take = [&](size_t bytes) -> char* {
    char* p = ws + off;
    off = (off + bytes + 255) & ~(size_t)255;
    return p;
  };
  unsigned short* xpadT = (unsigned short*)take((size_t)B_ * PW_ * PW_ * CIN_ * 2);
  float*          scl   = (float*)take(768 * 4);
  unsigned short* Wfq   = (unsigned short*)take((size_t)256 * KC_ * 2);
  unsigned short* Wfk   = (unsigned short*)take((size_t)256 * KC_ * 2);
  unsigned short* Wfv   = (unsigned short*)take((size_t)256 * KC_ * 2);
  float* qbuf = (float*)take((size_t)B_ * S_ * 256 * 4);
  float* kbuf = (float*)take((size_t)B_ * S_ * 256 * 4);
  float* vbuf = (float*)take((size_t)B_ * S_ * 256 * 4);

  scale_kernel<<<3, 256, 0, stream>>>(q_v, q_g, k_v, k_g, v_v, v_g, scl);

  int fuse_blocks = (256 * KC_) / 256;
  fuse_qk_kernel<<<fuse_blocks, 256, 0, stream>>>(q_v, scl + 0,   wq, 512, Wfq);
  fuse_qk_kernel<<<fuse_blocks, 256, 0, stream>>>(k_v, scl + 256, wk, 256, Wfk);
  fuse_v_kernel <<<fuse_blocks, 256, 0, stream>>>(v_v, scl + 512, wv, Wfv);

  int pad_total = B_ * PW_ * PW_ * CIN_;
  pad_kernel<<<(pad_total + 255) / 256, 256, 0, stream>>>(left, right, xpadT);

  dim3 cgrid(128, 4);  // 128 N tiles (b, 4-row group) x 4 M tiles
  conv_wmma_kernel<<<cgrid, 256, 0, stream>>>(Wfq, xpadT, q_b, qbuf);
  conv_wmma_kernel<<<cgrid, 256, 0, stream>>>(Wfk, xpadT, k_b, kbuf);
  conv_wmma_kernel<<<cgrid, 256, 0, stream>>>(Wfv, xpadT, v_b, vbuf);

  attn_kernel<<<(B_ * S_ * NH_) / 256, 256, 0, stream>>>(qbuf, kbuf, vbuf, out);
}
